// SetTransformer_66400194396709
// MI455X (gfx1250) — matrix operand order repaired and checked by reading, not compiled, not run on hardware
//
#include <hip/hip_runtime.h>
#include <hip/hip_bf16.h>
#include <stdint.h>

// Problem dims (fixed by the reference)
#define B_    32
#define N_    4096
#define DIN_  64
#define DH_   256
#define NH_   4
#define HD_   64      // DH_/NH_
#define NI_   32
#define NOUT_ 1
#define DOUT_ 16

typedef __attribute__((ext_vector_type(16))) _Float16 v16h;
typedef __attribute__((ext_vector_type(8)))  float    v8f;
typedef __attribute__((ext_vector_type(4)))  int      v4i;

union FragH { v16h v; _Float16 h[16]; };
union FragF { v8f  v; float    f[8];  };
union H8    { uint4 q; _Float16 h[8]; };

// CDNA5 async global->LDS copy (ASYNCcnt path), if the toolchain exposes it.
#if __has_builtin(__builtin_amdgcn_global_load_async_to_lds_b128) && \
    __has_builtin(__builtin_amdgcn_s_wait_asynccnt)
#define ASYNC_OK 1
#else
#define ASYNC_OK 0
#endif

__device__ __forceinline__ void gl2lds16(const _Float16* g, _Float16* l) {
#if ASYNC_OK
  // Builtin signature: (v4i addrspace(1)*, v4i addrspace(3)*, imm offset, imm cpol).
  // Generic->LDS: hardware uses addr[31:0] as the LDS offset, so the low 32
  // bits of the generic pointer are the addrspace(3) value.
  __builtin_amdgcn_global_load_async_to_lds_b128(
      (__attribute__((address_space(1))) v4i*)(uintptr_t)g,
      (__attribute__((address_space(3))) v4i*)(unsigned)(uintptr_t)l,
      0, 0);
#else
  *(uint4*)l = *(const uint4*)g;
#endif
}

template <int NCNT>
__device__ __forceinline__ void asyncWait() {
#if ASYNC_OK
  __builtin_amdgcn_s_wait_asynccnt(NCNT);
#endif
}

// ---------------------------------------------------------------------------
// Generic batched WMMA GEMM:  C[z] = epilogue( alpha * A[z] x B[z] + bias )
//   z = b*nh + h ; per-operand (b,h) pointer strides allow head slicing and
//   broadcast (stride 0). B addressed as B[k*ldbk + n*ldbn] (handles k^T).
//   BMODE: 0 = B contiguous along k (ldbk==1), 1 = B contiguous along n.
//   EXACT: M%TM==0 && N%TN==0 && K%TK==0 && aligned -> no guards, and the
//          K-loop is double-buffered: async copies for tile i+1 are issued
//          before waiting on tile i (ASYNCcnt completes in order), so global
//          latency hides behind the WMMA work.
// ---------------------------------------------------------------------------
struct GemmP {
  const _Float16* A;
  const _Float16* Bm;
  const float*    bias;
  float*          C;
  _Float16*       C16;
  const float*    D;          // residual add source (f32)
  int M, N, K;
  int lda, ldbk, ldbn, ldc, ldd;
  long long sAb, sAh, sBb, sBh, sCb, sCh, sDb, sDh;
  int   nh;
  float alpha;
  int   relu;
};

#define TM   64
#define TN   64
#define TK   32
#define LDSW 40   // padded row stride in halves (80B: 16B aligned, conflict-skewed)

template <int BMODE, bool EXACT>
__global__ __launch_bounds__(256)
void k_gemm_wmma(GemmP p) {
  __shared__ __align__(16) _Float16 Al[2][TM * LDSW];
  __shared__ __align__(16) _Float16 Bl[2][TN * LDSW];

  const int tid  = threadIdx.x;
  const int lane = tid & 31;
  const int wave = tid >> 5;      // 8 waves
  const int wm   = wave >> 1;     // 4 row tiles
  const int wn   = wave & 1;      // 2 col strips (each 32 wide)
  const int gm0  = blockIdx.y * TM;
  const int gn0  = blockIdx.x * TN;
  const int z    = blockIdx.z;
  const int bb   = z / p.nh;
  const int hh   = z - bb * p.nh;

  const _Float16* __restrict__ A  = p.A  + bb * p.sAb + hh * p.sAh;
  const _Float16* __restrict__ Bm = p.Bm + bb * p.sBb + hh * p.sBh;

  FragF acc0, acc1;
#pragma unroll
  for (int j = 0; j < 8; ++j) { acc0.f[j] = 0.f; acc1.f[j] = 0.f; }

  const int sr = tid >> 2;        // 0..63 : tile row (A) / tile col (B, k-contig)
  const int sc = (tid & 3) * 8;   // 0,8,16,24 : K sub-range
  // mapping for n-contiguous B staging
  const int kIdx = tid >> 3;      // 0..31
  const int n0   = (tid & 7) * 8; // 0..56

  // Fragment compute on one LDS buffer pair.
  auto computeTile = [&](const _Float16* Abuf, const _Float16* Bbuf) {
    FragH fa, fb0, fb1;
    const int r  = wm * 16 + (lane & 15);
    const int kh = (lane >> 4) * 8;          // lanes 0-15: K 0-7/16-23 ; 16-31: K 8-15/24-31
    *(uint4*)&fa.h[0] = *(const uint4*)&Abuf[r * LDSW + kh];
    *(uint4*)&fa.h[8] = *(const uint4*)&Abuf[r * LDSW + 16 + kh];
    const int c0 = wn * 32 + (lane & 15);
    const int kb = (lane >> 4) * 8;          // lanes 0-15: K 0-7/16-23 ; 16-31: K 8-15/24-31
    *(uint4*)&fb0.h[0] = *(const uint4*)&Bbuf[c0 * LDSW + kb];
    *(uint4*)&fb0.h[8] = *(const uint4*)&Bbuf[c0 * LDSW + kb + 16];
    const int c1 = c0 + 16;
    *(uint4*)&fb1.h[0] = *(const uint4*)&Bbuf[c1 * LDSW + kb];
    *(uint4*)&fb1.h[8] = *(const uint4*)&Bbuf[c1 * LDSW + kb + 16];
    acc0.v = __builtin_amdgcn_wmma_f32_16x16x32_f16(false, fa.v, false, fb0.v,
                                                    (short)0, acc0.v, false, false);
    acc1.v = __builtin_amdgcn_wmma_f32_16x16x32_f16(false, fa.v, false, fb1.v,
                                                    (short)0, acc1.v, false, false);
  };

  if (EXACT) {
    // ---- double-buffered async pipeline (no guards needed) ----
    auto stage = [&](int kk, _Float16* Abuf, _Float16* Bbuf) {
      gl2lds16(&A[(long long)(gm0 + sr) * p.lda + kk + sc],
               &Abuf[sr * LDSW + sc]);
      if (BMODE == 0) {
        gl2lds16(&Bm[(long long)(gn0 + sr) * p.ldbn + kk + sc],
                 &Bbuf[sr * LDSW + sc]);
      } else {
        H8 t;
        t.q = *(const uint4*)&Bm[(long long)(kk + kIdx) * p.ldbk + gn0 + n0];
#pragma unroll
        for (int e = 0; e < 8; ++e)
          Bbuf[(n0 + e) * LDSW + kIdx] = t.h[e];
      }
    };

    const int nTiles = p.K / TK;
    stage(0, Al[0], Bl[0]);
    for (int i = 0; i < nTiles; ++i) {
      const int cur = i & 1;
      if (i + 1 < nTiles) {
        stage((i + 1) * TK, Al[1 - cur], Bl[1 - cur]);
        // tile i's copies complete before tile i+1's (in-order ASYNCcnt):
        asyncWait<(BMODE == 0) ? 2 : 1>();
      } else {
        asyncWait<0>();
      }
      __syncthreads();                 // everyone's buffer `cur` is resident
      computeTile(Al[cur], Bl[cur]);
      __syncthreads();                 // reads done before `cur` is restaged
    }
  } else {
    // ---- guarded single-buffer loop (branch-free clamped fallbacks) ----
    for (int kk = 0; kk < p.K; kk += TK) {
      {
        const int  grow    = gm0 + sr;
        const long long rb = (long long)(grow < p.M ? grow : (p.M - 1)) * p.lda;
        if (((p.lda & 7) == 0) && grow < p.M && (kk + TK) <= p.K) {
          *(uint4*)&Al[0][sr * LDSW + sc] = *(const uint4*)&A[rb + kk + sc];
          if (kk + TK < p.K)
            __builtin_prefetch(&A[rb + kk + TK + sc], 0, 0);
        } else {
          H8 t;
#pragma unroll
          for (int e = 0; e < 8; ++e) {
            const int gk = kk + sc + e;
            const _Float16 v = A[rb + (gk < p.K ? gk : 0)];   // clamped, valid
            t.h[e] = (grow < p.M && gk < p.K) ? v : (_Float16)0.f;
          }
          *(uint4*)&Al[0][sr * LDSW + sc] = t.q;
        }
      }
      if (BMODE == 0) {
        const int  gn      = gn0 + sr;
        const long long cb = (long long)(gn < p.N ? gn : (p.N - 1)) * p.ldbn;
        if (((p.ldbn & 7) == 0) && gn < p.N && (kk + TK) <= p.K) {
          *(uint4*)&Bl[0][sr * LDSW + sc] = *(const uint4*)&Bm[cb + kk + sc];
        } else {
          H8 t;
#pragma unroll
          for (int e = 0; e < 8; ++e) {
            const int gk = kk + sc + e;
            const _Float16 v = Bm[cb + (gk < p.K ? gk : 0)];
            t.h[e] = (gn < p.N && gk < p.K) ? v : (_Float16)0.f;
          }
          *(uint4*)&Bl[0][sr * LDSW + sc] = t.q;
        }
      } else {
        const int  gk      = kk + kIdx;
        const long long kb = (long long)(gk < p.K ? gk : 0) * p.ldbk;
        if (((p.ldbk & 7) == 0) && gk < p.K && (gn0 + n0 + 8) <= p.N) {
          H8 t;
          t.q = *(const uint4*)&Bm[kb + gn0 + n0];
#pragma unroll
          for (int e = 0; e < 8; ++e)
            Bl[0][(n0 + e) * LDSW + kIdx] = t.h[e];
        } else {
#pragma unroll
          for (int e = 0; e < 8; ++e) {
            const int gn = gn0 + n0 + e;
            const long long nb = (long long)(gn < p.N ? gn : 0) * p.ldbn;
            const _Float16 v = Bm[kb + nb];                   // clamped, valid
            Bl[0][(n0 + e) * LDSW + kIdx] =
                (gk < p.K && gn < p.N) ? v : (_Float16)0.f;
          }
        }
      }
      __syncthreads();
      computeTile(Al[0], Bl[0]);
      __syncthreads();
    }
  }

  float*       C   = p.C   ? p.C   + bb * p.sCb + hh * p.sCh : nullptr;
  _Float16*    C16 = p.C16 ? p.C16 + bb * p.sCb + hh * p.sCh : nullptr;
  const float* D   = p.D   ? p.D   + bb * p.sDb + hh * p.sDh : nullptr;

#pragma unroll
  for (int t = 0; t < 2; ++t) {
    const FragF& a   = t ? acc1 : acc0;
    const int    col = gn0 + wn * 32 + t * 16 + (lane & 15);
    if (!EXACT && col >= p.N) continue;
    const float bv = p.bias ? p.bias[col] : 0.f;
#pragma unroll
    for (int j = 0; j < 8; ++j) {
      const int row = gm0 + wm * 16 + (lane >> 4) * 8 + j;  // C layout: VGPR j -> M=j (+8 hi lanes)
      if (!EXACT && row >= p.M) continue;
      float v = p.alpha * a.f[j] + bv;
      if (p.relu) v = v > 0.f ? v : 0.f;
      if (D) v += D[(long long)row * p.ldd + col];
      if (C)   C  [(long long)row * p.ldc + col] = v;
      if (C16) C16[(long long)row * p.ldc + col] = (_Float16)v;
    }
  }
}

// ---------------------------------------------------------------------------
// f32 -> f16 conversion
// ---------------------------------------------------------------------------
__global__ void k_f32_to_f16(const float* __restrict__ s, _Float16* __restrict__ d,
                             long long n) {
  long long i = (long long)blockIdx.x * blockDim.x + threadIdx.x;
  if (i < n) d[i] = (_Float16)s[i];
}

// ---------------------------------------------------------------------------
// Row softmax with optional key-padding mask (mask==true -> -inf).
// row = (b*nh + h)*nq + q ; writes probabilities directly in f16.
// ---------------------------------------------------------------------------
__global__ __launch_bounds__(128)
void k_softmax(const float* __restrict__ S, _Float16* __restrict__ P,
               const unsigned char* __restrict__ mask,
               int nk, int nq, int nh, int maskN) {
  const int row = blockIdx.x;
  const int b   = row / (nh * nq);
  const float* s = S + (long long)row * nk;
  _Float16*    o = P + (long long)row * nk;
  const unsigned char* mrow = mask ? mask + (long long)b * maskN : nullptr;
  __shared__ float red[128];
  const int t = threadIdx.x;

  float m = -3.4e38f;
  for (int k = t; k < nk; k += 128) {
    float v = s[k];
    if (mrow && mrow[k]) v = -3.4e38f;
    m = fmaxf(m, v);
  }
  red[t] = m; __syncthreads();
  for (int oo = 64; oo > 0; oo >>= 1) { if (t < oo) red[t] = fmaxf(red[t], red[t + oo]); __syncthreads(); }
  m = red[0]; __syncthreads();

  float sum = 0.f;
  for (int k = t; k < nk; k += 128) {
    float v = s[k];
    if (mrow && mrow[k]) v = -3.4e38f;
    sum += __expf(v - m);
  }
  red[t] = sum; __syncthreads();
  for (int oo = 64; oo > 0; oo >>= 1) { if (t < oo) red[t] += red[t + oo]; __syncthreads(); }
  const float inv = 1.f / red[0];
  __syncthreads();

  for (int k = t; k < nk; k += 128) {
    float v = s[k];
    if (mrow && mrow[k]) v = -3.4e38f;
    o[k] = (_Float16)(__expf(v - m) * inv);
  }
}

// ---------------------------------------------------------------------------
// LayerNorm over the last dim (DH_=256), one row per 256-thread block.
// Writes f32 and/or f16.
// ---------------------------------------------------------------------------
__global__ __launch_bounds__(256)
void k_layernorm(const float* __restrict__ X, const float* __restrict__ g,
                 const float* __restrict__ bb, float* __restrict__ Y32,
                 _Float16* __restrict__ Y16) {
  const int row = blockIdx.x;
  const int t   = threadIdx.x;
  const float v = X[(long long)row * DH_ + t];
  __shared__ float red[256];
  red[t] = v; __syncthreads();
  for (int o = 128; o > 0; o >>= 1) { if (t < o) red[t] += red[t + o]; __syncthreads(); }
  const float mu = red[0] * (1.f / DH_); __syncthreads();
  const float d  = v - mu;
  red[t] = d * d; __syncthreads();
  for (int o = 128; o > 0; o >>= 1) { if (t < o) red[t] += red[t + o]; __syncthreads(); }
  const float var = red[0] * (1.f / DH_);
  const float y   = d * rsqrtf(var + 1e-5f) * g[t] + bb[t];
  if (Y32) Y32[(long long)row * DH_ + t] = y;
  if (Y16) Y16[(long long)row * DH_ + t] = (_Float16)y;
}

// ---------------------------------------------------------------------------
// Host orchestration
// ---------------------------------------------------------------------------
static inline size_t alignUp256(size_t x) { return (x + 255) & ~(size_t)255; }

extern "C" void kernel_launch(void* const* d_in, const int* in_sizes, int n_in,
                              void* d_out, int out_size, void* d_ws, size_t ws_size,
                              hipStream_t stream) {
  (void)in_sizes; (void)n_in; (void)out_size; (void)ws_size;

  // Input order: X, mask, then params pytree flattened in sorted-key order.
  const float*         X    = (const float*)d_in[0];
  const unsigned char* mask = (const unsigned char*)d_in[1];  // jnp bool_ = 1 byte
  const float* I1   = (const float*)d_in[2];
  const float* I2   = (const float*)d_in[3];
  const float* Sv   = (const float*)d_in[4];
  const float* fcW  = (const float*)d_in[5];
  const float* fcb  = (const float*)d_in[6];

  struct MabW { const float *Wk,*Wo,*Wq,*Wv,*b0,*b1,*bk,*bo,*bq,*bv,*g0,*g1; };
  auto mabAt = [&](int base) -> MabW {
    MabW m;
    m.Wk = (const float*)d_in[base + 0];  m.Wo = (const float*)d_in[base + 1];
    m.Wq = (const float*)d_in[base + 2];  m.Wv = (const float*)d_in[base + 3];
    m.b0 = (const float*)d_in[base + 4];  m.b1 = (const float*)d_in[base + 5];
    m.bk = (const float*)d_in[base + 6];  m.bo = (const float*)d_in[base + 7];
    m.bq = (const float*)d_in[base + 8];  m.bv = (const float*)d_in[base + 9];
    m.g0 = (const float*)d_in[base + 10]; m.g1 = (const float*)d_in[base + 11];
    return m;
  };
  const MabW m10 = mabAt(7),  m11 = mabAt(19), m20 = mabAt(31), m21 = mabAt(43);
  const MabW mpm = mabAt(55), ms1 = mabAt(67), ms2 = mabAt(79);

  // ---- workspace bump allocator ----
  char*  wsbase = (char*)d_ws;
  size_t off = 0;
  auto alloc = [&](size_t bytes) -> void* {
    void* p = (void*)(wsbase + off);
    off = alignUp256(off + bytes);
    return p;
  };

  const long long actH = (long long)B_ * N_ * DH_;  // 33.5M elements

  _Float16* curA = (_Float16*)alloc((size_t)actH * 2);               // X activation ping
  _Float16* curB = (_Float16*)alloc((size_t)actH * 2);               // X activation pong
  _Float16* H16  = (_Float16*)alloc((size_t)B_ * NI_ * DH_ * 2);     // inducing summary
  _Float16* smA  = (_Float16*)alloc((size_t)B_ * NOUT_ * DH_ * 2);   // pooled ping
  _Float16* smB  = (_Float16*)alloc((size_t)B_ * NOUT_ * DH_ * 2);   // pooled pong
  float*    q32  = (float*)   alloc((size_t)actH * 4);  // q proj f32; reused as LN0 f32
  _Float16* q16  = (_Float16*)alloc((size_t)actH * 2);  // q proj f16; reused as LN0 f16
  _Float16* k16  = (_Float16*)alloc((size_t)actH * 2);
  _Float16* v16  = (_Float16*)alloc((size_t)actH * 2);
  float*    o32  = (float*)   alloc((size_t)actH * 4);  // attn-out / Wo-out f32
  float*    s32  = o32;                                 // scores alias (disjoint lifetime)
  _Float16* p16  = (_Float16*)alloc((size_t)B_ * NH_ * (size_t)N_ * NI_ * 2);

  auto cvt = [&](const float* src, long long n) -> _Float16* {
    _Float16* dst = (_Float16*)alloc((size_t)n * 2);
    const unsigned blocks = (unsigned)((n + 255) / 256);
    k_f32_to_f16<<<dim3(blocks), dim3(256), 0, stream>>>(src, dst, n);
    return dst;
  };

  _Float16* x16   = cvt(X,   (long long)B_ * N_ * DIN_);
  _Float16* I116  = cvt(I1,  (long long)NI_ * DH_);
  _Float16* I216  = cvt(I2,  (long long)NI_ * DH_);
  _Float16* S16   = cvt(Sv,  (long long)NOUT_ * DH_);
  _Float16* fcW16 = cvt(fcW, (long long)DH_ * DOUT_);

  struct W16 { _Float16 *Wq, *Wk, *Wv, *Wo; };
  auto cvtW = [&](const MabW& m, int dq, int dk) -> W16 {
    W16 r;
    r.Wq = cvt(m.Wq, (long long)dq * DH_);
    r.Wk = cvt(m.Wk, (long long)dk * DH_);
    r.Wv = cvt(m.Wv, (long long)dk * DH_);
    r.Wo = cvt(m.Wo, (long long)DH_ * DH_);
    return r;
  };
  const W16 w10 = cvtW(m10, DH_, DIN_);
  const W16 w11 = cvtW(m11, DIN_, DH_);
  const W16 w20 = cvtW(m20, DH_, DH_);
  const W16 w21 = cvtW(m21, DH_, DH_);
  const W16 wpm = cvtW(mpm, DH_, DH_);
  const W16 wsa = cvtW(ms1, DH_, DH_);
  const W16 wsc = cvtW(ms2, DH_, DH_);

  // Dispatch to the right template instantiation.
  auto gemm = [&](const GemmP& p, int batch) {
    dim3 grid((p.N + TN - 1) / TN, (p.M + TM - 1) / TM, batch);
    const bool bk = (p.ldbk == 1);
    const bool exact = (p.M % TM == 0) && (p.N % TN == 0) && (p.K % TK == 0) &&
                       ((p.lda & 7) == 0) &&
                       (bk ? ((p.ldbn & 7) == 0)
                           : ((p.ldbn == 1) && ((p.ldbk & 7) == 0)));
    if (bk) {
      if (exact) k_gemm_wmma<0, true ><<<grid, dim3(256), 0, stream>>>(p);
      else       k_gemm_wmma<0, false><<<grid, dim3(256), 0, stream>>>(p);
    } else {
      if (exact) k_gemm_wmma<1, true ><<<grid, dim3(256), 0, stream>>>(p);
      else       k_gemm_wmma<1, false><<<grid, dim3(256), 0, stream>>>(p);
    }
  };

  // Full MAB: q/k/v proj -> scores -> softmax -> attn+residual -> LN -> Wo -> LN
  auto runMab = [&](const _Float16* Qin, int dq, int Nq, bool qb,
                    const _Float16* Kin, int dk, int Nk,
                    const MabW& wf, const W16& wh,
                    const unsigned char* mk, _Float16* out16) {
    const int Mq = qb ? Nq : B_ * Nq;
    {   // q = Q @ Wq + bq  (f32 + f16)
      GemmP p{};
      p.A = Qin; p.Bm = wh.Wq; p.bias = wf.bq; p.C = q32; p.C16 = q16;
      p.M = Mq; p.N = DH_; p.K = dq;
      p.lda = dq; p.ldbk = DH_; p.ldbn = 1; p.ldc = DH_;
      p.nh = 1; p.alpha = 1.f;
      gemm(p, 1);
    }
    {   // k = K @ Wk + bk  (f16)
      GemmP p{};
      p.A = Kin; p.Bm = wh.Wk; p.bias = wf.bk; p.C16 = k16;
      p.M = B_ * Nk; p.N = DH_; p.K = dk;
      p.lda = dk; p.ldbk = DH_; p.ldbn = 1; p.ldc = DH_;
      p.nh = 1; p.alpha = 1.f;
      gemm(p, 1);
    }
    {   // v = K @ Wv + bv  (f16)
      GemmP p{};
      p.A = Kin; p.Bm = wh.Wv; p.bias = wf.bv; p.C16 = v16;
      p.M = B_ * Nk; p.N = DH_; p.K = dk;
      p.lda = dk; p.ldbk = DH_; p.ldbn = 1; p.ldc = DH_;
      p.nh = 1; p.alpha = 1.f;
      gemm(p, 1);
    }
    {   // s[b,h] = q_h @ k_h^T / sqrt(DH)
      GemmP p{};
      p.A = q16; p.Bm = k16; p.C = s32;
      p.M = Nq; p.N = Nk; p.K = HD_;
      p.lda = DH_; p.ldbk = 1; p.ldbn = DH_; p.ldc = Nk;
      p.sAb = qb ? 0 : (long long)Nq * DH_; p.sAh = HD_;
      p.sBb = (long long)Nk * DH_;          p.sBh = HD_;
      p.sCb = (long long)NH_ * Nq * Nk;     p.sCh = (long long)Nq * Nk;
      p.nh = NH_; p.alpha = 1.f / 16.f;     // 1/sqrt(256)
      gemm(p, B_ * NH_);
    }
    // softmax (masked for ISAB1 first MAB), probs -> f16
    k_softmax<<<dim3(B_ * NH_ * Nq), dim3(128), 0, stream>>>(
        s32, p16, mk, Nk, Nq, NH_, N_);
    {   // o = qh + A @ v_h   (overwrites dead scores region)
      GemmP p{};
      p.A = p16; p.Bm = v16; p.C = o32; p.D = q32;
      p.M = Nq; p.N = HD_; p.K = Nk;
      p.lda = Nk; p.ldbk = DH_; p.ldbn = 1; p.ldc = DH_; p.ldd = DH_;
      p.sAb = (long long)NH_ * Nq * Nk;     p.sAh = (long long)Nq * Nk;
      p.sBb = (long long)Nk * DH_;          p.sBh = HD_;
      p.sCb = (long long)Nq * DH_;          p.sCh = HD_;
      p.sDb = qb ? 0 : (long long)Nq * DH_; p.sDh = HD_;
      p.nh = NH_; p.alpha = 1.f;
      gemm(p, B_ * NH_);
    }
    // LN0: o32 -> (q32, q16)  [q buffers are dead, reuse them]
    k_layernorm<<<dim3(B_ * Nq), dim3(256), 0, stream>>>(o32, wf.g0, wf.b0, q32, q16);
    {   // o = ln0 + relu(ln0 @ Wo + bo)
      GemmP p{};
      p.A = q16; p.Bm = wh.Wo; p.bias = wf.bo; p.C = o32; p.D = q32;
      p.M = B_ * Nq; p.N = DH_; p.K = DH_;
      p.lda = DH_; p.ldbk = DH_; p.ldbn = 1; p.ldc = DH_; p.ldd = DH_;
      p.nh = 1; p.alpha = 1.f; p.relu = 1;
      gemm(p, 1);
    }
    // LN1 -> out16
    k_layernorm<<<dim3(B_ * Nq), dim3(256), 0, stream>>>(o32, wf.g1, wf.b1,
                                                         nullptr, out16);
  };

  // ---- forward pass ----
  runMab(I116, DH_, NI_,   true,  x16,  DIN_, N_,    m10, w10, mask,    H16);  // ISAB1.0 (masked)
  runMab(x16,  DIN_, N_,   false, H16,  DH_,  NI_,   m11, w11, nullptr, curA); // ISAB1.1
  runMab(I216, DH_, NI_,   true,  curA, DH_,  N_,    m20, w20, nullptr, H16);  // ISAB2.0
  runMab(curA, DH_, N_,    false, H16,  DH_,  NI_,   m21, w21, nullptr, curB); // ISAB2.1
  runMab(S16,  DH_, NOUT_, true,  curB, DH_,  N_,    mpm, wpm, nullptr, smA);  // PMA
  runMab(smA,  DH_, NOUT_, false, smA,  DH_,  NOUT_, ms1, wsa, nullptr, smB);  // SAB1
  runMab(smB,  DH_, NOUT_, false, smB,  DH_,  NOUT_, ms2, wsc, nullptr, smA);  // SAB2

  {   // final linear: [B,256] @ [256,16] + fcb -> d_out (f32)
    GemmP p{};
    p.A = smA; p.Bm = fcW16; p.bias = fcb; p.C = (float*)d_out;
    p.M = B_; p.N = DOUT_; p.K = DH_;
    p.lda = DH_; p.ldbk = DOUT_; p.ldbn = 1; p.ldc = DOUT_;
    p.nh = 1; p.alpha = 1.f;
    gemm(p, 1);
  }
}
